// SS3D_69355131896212
// MI455X (gfx1250) — matrix operand order repaired and checked by reading, not compiled, not run on hardware
//
#include <hip/hip_runtime.h>
#include <hip/hip_bf16.h>

// ---------------------------------------------------------------------------
// SS3D (VMamba-style selective scan) for MI455X / gfx1250, wave32 + WMMA.
// Shapes: B=2, H=W=64 (L=4096), DM=64, DI=128, N=16, R=4, K=8 (only k=0..3 live)
// ---------------------------------------------------------------------------

#define B_ 2
#define L_ 4096
#define DM_ 64
#define DI_ 128
#define N_ 16
#define KLIVE 4          // k_sel reuses directions 0..3; 4..7 outputs are dead
#define CHUNK 64         // scan chunk length
#define NCHUNK (L_ / CHUNK)

typedef __attribute__((ext_vector_type(16))) __bf16 v16bf;
typedef __attribute__((ext_vector_type(8)))  float  v8f;

union BfFrag { v16bf v; __bf16 h[16]; };

#if defined(__gfx1250__) && __has_builtin(__builtin_amdgcn_global_load_async_to_lds_b32) && \
    __has_builtin(__builtin_amdgcn_s_wait_asynccnt)
#define HAS_ASYNC_LDS 1
#else
#define HAS_ASYNC_LDS 0
#endif

// A-fragment (16x32, 16-bit): lane half hi: K = hi*16 + e
__device__ __forceinline__ int a_k(int e, int hi) {
  return hi * 16 + e;
}
// B-fragment (32x16, 16-bit): lanes 0-15 hold K=0..15, lanes 16-31 hold K=16..31
__device__ __forceinline__ int b_k(int e, int hi) { return hi * 16 + e; }

__device__ __forceinline__ v8f wmma_bf16(v16bf a, v16bf b, v8f c) {
  return __builtin_amdgcn_wmma_f32_16x16x32_bf16(
      /*neg_a=*/false, a, /*neg_b=*/false, b,
      /*c_mod=*/(short)0, c, /*reuse_a=*/false, /*reuse_b=*/false);
}

__device__ __forceinline__ float softplusf(float s) {
  return (s > 20.f) ? s : log1pf(__expf(s));
}
__device__ __forceinline__ float silu_f(float s) {
  return s / (1.f + __expf(-s));
}

// stage one f32 value into LDS, via async-copy engine when available
__device__ __forceinline__ void stage_lds(float* shp, const float* gp) {
#if HAS_ASYNC_LDS
  __builtin_amdgcn_global_load_async_to_lds_b32(
      (__attribute__((address_space(1))) int*)(gp),
      (__attribute__((address_space(3))) int*)(shp), 0, 0);
#else
  *shp = *gp;
#endif
}
__device__ __forceinline__ void stage_wait() {
#if HAS_ASYNC_LDS
  __builtin_amdgcn_s_wait_asynccnt(0);
#endif
}

// ---------------------------------------------------------------------------
// Kernel A: fused reverse3d + in_proj GEMM.  xz[m,n] = sum_c x_rev[m,c]*W[n,c]
// M = B*L = 8192, Kdim = 64, Ncols = 256 -> xin (first 128), z (last 128)
// grid (128, 16), block 128 = 4 waves, each wave one 16x16 tile
// ---------------------------------------------------------------------------
__global__ void __launch_bounds__(128)
k_inproj(const float* __restrict__ x, const float* __restrict__ w,
         float* __restrict__ xin, float* __restrict__ z) {
  const int lane = threadIdx.x & 31, hi = lane >> 4, lr = lane & 15;
  const int wv = threadIdx.x >> 5;
  const int m0 = (blockIdx.x * 4 + wv) * 16;
  const int n0 = blockIdx.y * 16;

  const int rowA = m0 + lr;
  const int bA = rowA >> 12, lA = rowA & (L_ - 1);
  const int flip = (lA + bA) & 1;
  const float* xrow = x + (long)rowA * DM_;
  const float* wrow = w + (long)(n0 + lr) * DM_;

  v8f acc = {};
#pragma unroll
  for (int kk = 0; kk < DM_; kk += 32) {
    BfFrag a, bb;
#pragma unroll
    for (int e = 0; e < 16; ++e) {
      int ca = kk + a_k(e, hi);
      int cs = flip ? (DM_ - 1 - ca) : ca;
      a.h[e] = (__bf16)xrow[cs];
      bb.h[e] = (__bf16)wrow[kk + b_k(e, hi)];
    }
    acc = wmma_bf16(a.v, bb.v, acc);
  }
  // tile is entirely in xin or entirely in z: uniform (scalar) select
  float* dstbase;
  int ncol0;
  if (n0 < DI_) { dstbase = xin; ncol0 = n0; }
  else          { dstbase = z;   ncol0 = n0 - DI_; }
#pragma unroll
  for (int r = 0; r < 8; ++r) {
    const int row = m0 + r + hi * 8;
    dstbase[(long)row * DI_ + ncol0 + lr] = acc[r];
  }
}

// ---------------------------------------------------------------------------
// Kernel B: depthwise conv1d (kernel 3, pad 1) over L + bias + SiLU.
// reads xin [B,L,Di], writes xc [B,Di,L]
// ---------------------------------------------------------------------------
__global__ void __launch_bounds__(256)
k_conv(const float* __restrict__ xin, const float* __restrict__ cw,
       const float* __restrict__ cb, float* __restrict__ xc) {
  int idx = blockIdx.x * blockDim.x + threadIdx.x;   // b*Di*L
  if (idx >= B_ * DI_ * L_) return;
  int l = idx & (L_ - 1);
  int d = (idx >> 12) & (DI_ - 1);
  int b = idx >> 19;
  const float* base = xin + (long)b * L_ * DI_ + d;
  float xm = (l > 0)      ? base[(long)(l - 1) * DI_] : 0.f;
  float x0 = base[(long)l * DI_];
  float xp = (l < L_ - 1) ? base[(long)(l + 1) * DI_] : 0.f;
  float s = cw[d * 3 + 0] * xm + cw[d * 3 + 1] * x0 + cw[d * 3 + 2] * xp + cb[d];
  xc[idx] = silu_f(s);
}

// ---------------------------------------------------------------------------
// Kernel C: x_dbl GEMM with fused zigzag gather (k = 0..3 only).
// x_dbl[b,k,c,l] = sum_d xpw[k,c,d] * xc[b,d,zig[k,l]]
// M = 36 (3 tiles of 16), Ncols = L, Kdim = 128. grid (64, 3, 8), block 128
// ---------------------------------------------------------------------------
__global__ void __launch_bounds__(128)
k_xdbl(const float* __restrict__ xc, const int* __restrict__ zig,
       const float* __restrict__ xpw, float* __restrict__ xdbl) {
  const int lane = threadIdx.x & 31, hi = lane >> 4, lr = lane & 15;
  const int wv = threadIdx.x >> 5;
  const int l0 = (blockIdx.x * 4 + wv) * 16;
  const int c0 = blockIdx.y * 16;
  const int bk = blockIdx.z, b = bk >> 2, k = bk & 3;

  const int l = l0 + lr;
  const int zp = zig[k * L_ + l];                    // gathered column index
  const float* xcb = xc + (long)b * DI_ * L_;
  const float* wA  = xpw + (long)k * 36 * DI_;
  const int rowc = c0 + lr;

  v8f acc = {};
#pragma unroll
  for (int kk = 0; kk < DI_; kk += 32) {
    BfFrag a, bb;
#pragma unroll
    for (int e = 0; e < 16; ++e) {
      int da = kk + a_k(e, hi);
      a.h[e] = (rowc < 36) ? (__bf16)wA[(long)rowc * DI_ + da] : (__bf16)0.f;
      int db = kk + b_k(e, hi);
      bb.h[e] = (__bf16)xcb[(long)db * L_ + zp];
    }
    acc = wmma_bf16(a.v, bb.v, acc);
  }
#pragma unroll
  for (int r = 0; r < 8; ++r) {
    int c = c0 + r + hi * 8;
    if (c < 36) xdbl[((long)bk * 36 + c) * L_ + l] = acc[r];
  }
}

// ---------------------------------------------------------------------------
// Scan pass 1: per chunk, per (b,k,d): decay product P[n] and zero-state S[n].
// delta computed on the fly from xdbl rows 0..3 (dt proj + softplus).
// grid (8, 64), block 128 (one thread per d channel)
// ---------------------------------------------------------------------------
__global__ void __launch_bounds__(128)
k_scan1(const float* __restrict__ xc, const int* __restrict__ zig,
        const float* __restrict__ xdbl, const float* __restrict__ dtw,
        const float* __restrict__ dtb, const float* __restrict__ Alog,
        float* __restrict__ P, float* __restrict__ S) {
  __shared__ float sh[36];
  const int bk = blockIdx.x, b = bk >> 2, k = bk & 3;
  const int j = blockIdx.y;
  const int d = threadIdx.x;

  float A[N_];
#pragma unroll
  for (int n = 0; n < N_; ++n) A[n] = -__expf(Alog[(k * DI_ + d) * N_ + n]);
  const float w0 = dtw[(k * DI_ + d) * 4 + 0], w1 = dtw[(k * DI_ + d) * 4 + 1];
  const float w2 = dtw[(k * DI_ + d) * 4 + 2], w3 = dtw[(k * DI_ + d) * 4 + 3];
  const float bias = dtb[k * DI_ + d];
  const float* xdb = xdbl + (long)bk * 36 * L_;
  const float* xcb = xc + (long)b * DI_ * L_ + (long)d * L_;
  const int* zgk = zig + k * L_;

  float p[N_], s[N_];
#pragma unroll
  for (int n = 0; n < N_; ++n) { p[n] = 1.f; s[n] = 0.f; }

  const int p0 = j * CHUNK;
  for (int t = 0; t < CHUNK; ++t) {
    const int pos = p0 + t;
    __syncthreads();
    if (threadIdx.x < 36) {
      stage_lds(&sh[threadIdx.x], &xdb[(long)threadIdx.x * L_ + pos]);
      if (pos + 4 < L_)
        __builtin_prefetch(&xdb[(long)threadIdx.x * L_ + pos + 4], 0, 1);
    }
    stage_wait();
    __syncthreads();
    const float delta = softplusf(w0 * sh[0] + w1 * sh[1] + w2 * sh[2] + w3 * sh[3] + bias);
    const float xv = xcb[zgk[pos]];
    const float du = delta * xv;
#pragma unroll
    for (int n = 0; n < N_; ++n) {
      const float dA = __expf(delta * A[n]);
      p[n] *= dA;
      s[n] = dA * s[n] + du * sh[4 + n];
    }
  }
  float* Pp = P + ((long)(bk * NCHUNK + j) * DI_ + d) * N_;
  float* Sp = S + ((long)(bk * NCHUNK + j) * DI_ + d) * N_;
#pragma unroll
  for (int n = 0; n < N_; ++n) { Pp[n] = p[n]; Sp[n] = s[n]; }
}

// ---------------------------------------------------------------------------
// Scan pass 2: propagate chunk states. 8*128*16 = 16384 independent chains.
// Hinit[bk,j,d,n] = state entering chunk j.
// ---------------------------------------------------------------------------
__global__ void __launch_bounds__(256)
k_scan2(const float* __restrict__ P, const float* __restrict__ S,
        float* __restrict__ Hinit) {
  int idx = blockIdx.x * blockDim.x + threadIdx.x;
  if (idx >= 8 * DI_ * N_) return;          // 8 = B_*KLIVE
  const int bk = idx >> 11;                 // / (128*16)
  const int dn = idx & 2047;
  const long base = (long)bk * NCHUNK * (DI_ * N_) + dn;
  float h = 0.f;
  for (int j = 0; j < NCHUNK; ++j) {
    const long o = base + (long)j * (DI_ * N_);
    Hinit[o] = h;
    h = P[o] * h + S[o];
  }
}

// ---------------------------------------------------------------------------
// Scan pass 3: replay each chunk from Hinit, emit out_y = y + Ds*xs.
// grid (8, 64), block 128
// ---------------------------------------------------------------------------
__global__ void __launch_bounds__(128)
k_scan3(const float* __restrict__ xc, const int* __restrict__ zig,
        const float* __restrict__ xdbl, const float* __restrict__ dtw,
        const float* __restrict__ dtb, const float* __restrict__ Alog,
        const float* __restrict__ Ds, const float* __restrict__ Hinit,
        float* __restrict__ outy) {
  __shared__ float sh[36];
  const int bk = blockIdx.x, b = bk >> 2, k = bk & 3;
  const int j = blockIdx.y;
  const int d = threadIdx.x;

  float A[N_];
#pragma unroll
  for (int n = 0; n < N_; ++n) A[n] = -__expf(Alog[(k * DI_ + d) * N_ + n]);
  const float w0 = dtw[(k * DI_ + d) * 4 + 0], w1 = dtw[(k * DI_ + d) * 4 + 1];
  const float w2 = dtw[(k * DI_ + d) * 4 + 2], w3 = dtw[(k * DI_ + d) * 4 + 3];
  const float bias = dtb[k * DI_ + d];
  const float Dd = Ds[k * DI_ + d];
  const float* xdb = xdbl + (long)bk * 36 * L_;
  const float* xcb = xc + (long)b * DI_ * L_ + (long)d * L_;
  const int* zgk = zig + k * L_;
  float* op = outy + ((long)bk * DI_ + d) * L_;

  float h[N_];
  const float* Hp = Hinit + ((long)(bk * NCHUNK + j) * DI_ + d) * N_;
#pragma unroll
  for (int n = 0; n < N_; ++n) h[n] = Hp[n];

  const int p0 = j * CHUNK;
  for (int t = 0; t < CHUNK; ++t) {
    const int pos = p0 + t;
    __syncthreads();
    if (threadIdx.x < 36) {
      stage_lds(&sh[threadIdx.x], &xdb[(long)threadIdx.x * L_ + pos]);
      if (pos + 4 < L_)
        __builtin_prefetch(&xdb[(long)threadIdx.x * L_ + pos + 4], 0, 1);
    }
    stage_wait();
    __syncthreads();
    const float delta = softplusf(w0 * sh[0] + w1 * sh[1] + w2 * sh[2] + w3 * sh[3] + bias);
    const float xv = xcb[zgk[pos]];
    const float du = delta * xv;
    float y = 0.f;
#pragma unroll
    for (int n = 0; n < N_; ++n) {
      const float dA = __expf(delta * A[n]);
      h[n] = dA * h[n] + du * sh[4 + n];
      y += h[n] * sh[20 + n];
    }
    op[pos] = y + Dd * xv;
  }
}

// ---------------------------------------------------------------------------
// Kernel F1: inverse zigzag merge (8 gathers, dirs 4..7 reuse out_y of 0..3)
// + LayerNorm over Di + SiLU(z) gate.  grid 8192 (one block per (b,l)), 128 thr
// ---------------------------------------------------------------------------
__global__ void __launch_bounds__(128)
k_merge_ln(const float* __restrict__ outy, const int* __restrict__ zb,
           const float* __restrict__ lnw, const float* __restrict__ lnb,
           const float* __restrict__ zg, float* __restrict__ yg) {
  const int bl = blockIdx.x;
  const int b = bl >> 12, l = bl & (L_ - 1);
  const int d = threadIdx.x;

  float acc = 0.f;
#pragma unroll
  for (int k = 0; k < KLIVE; ++k) {
    const int p1 = zb[k * L_ + l];
    const int p2 = zb[(k + 4) * L_ + l];
    const float* oy = outy + ((long)(b * KLIVE + k) * DI_ + d) * L_;
    acc += oy[p1] + oy[p2];
  }
  __shared__ float s1[DI_], s2[DI_];
  s1[d] = acc; s2[d] = acc * acc;
  __syncthreads();
  for (int off = DI_ / 2; off > 0; off >>= 1) {
    if (d < off) { s1[d] += s1[d + off]; s2[d] += s2[d + off]; }
    __syncthreads();
  }
  const float mu = s1[0] * (1.f / DI_);
  const float var = s2[0] * (1.f / DI_) - mu * mu;
  const float inv = rsqrtf(var + 1e-5f);
  const float yv = (acc - mu) * inv * lnw[d] + lnb[d];
  const float zv = zg[(long)bl * DI_ + d];
  yg[(long)bl * DI_ + d] = yv * silu_f(zv);
}

// ---------------------------------------------------------------------------
// Kernel F2: out_proj GEMM + fused output reverse3d.
// out[m,n] = sum_d yg[m,d]*opw[n,d]; M=8192, Kdim=128, Ncols=64
// grid (128, 4), block 128 = 4 waves
// ---------------------------------------------------------------------------
__global__ void __launch_bounds__(128)
k_outproj(const float* __restrict__ yg, const float* __restrict__ opw,
          float* __restrict__ out) {
  const int lane = threadIdx.x & 31, hi = lane >> 4, lr = lane & 15;
  const int wv = threadIdx.x >> 5;
  const int m0 = (blockIdx.x * 4 + wv) * 16;
  const int n0 = blockIdx.y * 16;
  const float* arow = yg + (long)(m0 + lr) * DI_;
  const float* wrow = opw + (long)(n0 + lr) * DI_;

  v8f acc = {};
#pragma unroll
  for (int kk = 0; kk < DI_; kk += 32) {
    BfFrag a, bb;
#pragma unroll
    for (int e = 0; e < 16; ++e) {
      a.h[e] = (__bf16)arow[kk + a_k(e, hi)];
      bb.h[e] = (__bf16)wrow[kk + b_k(e, hi)];
    }
    acc = wmma_bf16(a.v, bb.v, acc);
  }
#pragma unroll
  for (int r = 0; r < 8; ++r) {
    const int row = m0 + r + hi * 8;
    const int col = n0 + lr;
    const int bI = row >> 12, lI = row & (L_ - 1);
    const int flip = (lI + bI) & 1;
    const int cs = flip ? (DM_ - 1 - col) : col;
    out[(long)row * DM_ + cs] = acc[r];
  }
}

// ---------------------------------------------------------------------------
extern "C" void kernel_launch(void* const* d_in, const int* in_sizes, int n_in,
                              void* d_out, int out_size, void* d_ws, size_t ws_size,
                              hipStream_t stream) {
  const float* x    = (const float*)d_in[0];
  const int*   zig  = (const int*)  d_in[1];
  const int*   zb   = (const int*)  d_in[2];
  const float* ipw  = (const float*)d_in[3];
  const float* cw   = (const float*)d_in[4];
  const float* cb   = (const float*)d_in[5];
  const float* xpw  = (const float*)d_in[6];
  const float* dtw  = (const float*)d_in[7];
  const float* dtb  = (const float*)d_in[8];
  const float* Alog = (const float*)d_in[9];
  const float* Ds   = (const float*)d_in[10];
  const float* lnw  = (const float*)d_in[11];
  const float* lnb  = (const float*)d_in[12];
  const float* opw  = (const float*)d_in[13];
  float* out = (float*)d_out;

  // workspace carve-up (floats)
  float* ws   = (float*)d_ws;
  float* xin  = ws;                                   // B*L*DI   = 1048576
  float* zbuf = xin  + (long)B_ * L_ * DI_;           // 1048576
  float* xc   = zbuf + (long)B_ * L_ * DI_;           // 1048576
  float* xdbl = xc   + (long)B_ * L_ * DI_;           // 8*36*L   = 1179648
  float* P    = xdbl + (long)B_ * KLIVE * 36 * L_;    // 8*64*128*16 = 1048576
  float* S    = P    + (long)8 * NCHUNK * DI_ * N_;
  float* Hin  = S    + (long)8 * NCHUNK * DI_ * N_;
  float* outy = Hin  + (long)8 * NCHUNK * DI_ * N_;   // 8*128*L  = 4194304
  float* yg   = outy + (long)B_ * KLIVE * DI_ * L_;   // 1048576

  // 1) reverse3d + in_proj (WMMA bf16)
  k_inproj<<<dim3((B_ * L_) / 64, 256 / 16), 128, 0, stream>>>(x, ipw, xin, zbuf);
  // 2) depthwise conv + SiLU
  k_conv<<<(B_ * DI_ * L_ + 255) / 256, 256, 0, stream>>>(xin, cw, cb, xc);
  // 3) x_proj GEMM with fused zigzag gather (WMMA bf16), k = 0..3
  k_xdbl<<<dim3(L_ / 64, 3, B_ * KLIVE), 128, 0, stream>>>(xc, zig, xpw, xdbl);
  // 4) chunked selective scan (3 passes)
  k_scan1<<<dim3(B_ * KLIVE, NCHUNK), 128, 0, stream>>>(xc, zig, xdbl, dtw, dtb, Alog, P, S);
  k_scan2<<<(8 * DI_ * N_ + 255) / 256, 256, 0, stream>>>(P, S, Hin);
  k_scan3<<<dim3(B_ * KLIVE, NCHUNK), 128, 0, stream>>>(xc, zig, xdbl, dtw, dtb, Alog,
                                                        Ds, Hin, outy);
  // 5) inverse zigzag merge + LayerNorm + gate
  k_merge_ln<<<B_ * L_, 128, 0, stream>>>(outy, zb, lnw, lnb, zbuf, yg);
  // 6) out_proj (WMMA bf16) + reverse3d on output
  k_outproj<<<dim3((B_ * L_) / 64, DM_ / 16), 128, 0, stream>>>(yg, opw, out);

  (void)in_sizes; (void)n_in; (void)out_size; (void)ws_size;
}
